// AdvancedMoEMixtureLoRA_23854248362016
// MI455X (gfx1250) — compile-verified, matrix operand order checked by reading, not run on hardware
//
#include <hip/hip_runtime.h>
#include <stdint.h>

// ---------------------------------------------------------------------------
// Problem constants (match reference)
// ---------------------------------------------------------------------------
#define TOK   8192          // B*S tokens
#define IND   4096          // in_dim
#define OUTD  4096          // out_dim
#define NCOL  272           // 16 (A rows) + 256 (M rows)
#define KSPLIT 4
#define KSEG  (IND / KSPLIT)   // 1024

typedef __attribute__((ext_vector_type(16))) __bf16 v16bf;
typedef __attribute__((ext_vector_type(8)))  float  v8f;

union FragU { uint32_t u[8]; v16bf v; };

// fp32->bf16 pair pack (round-half-up): 2x v_add + 1x v_perm_b32
__device__ __forceinline__ uint32_t pkbf(float a, float b) {
    uint32_t ua = __float_as_uint(a) + 0x8000u;
    uint32_t ub = __float_as_uint(b) + 0x8000u;
    return __builtin_amdgcn_perm(ub, ua, 0x07060302u);  // {bf16(b)<<16 | bf16(a)}
}

__device__ __forceinline__ v8f wmma_bf16(const FragU& a, const FragU& b, v8f c) {
    return __builtin_amdgcn_wmma_f32_16x16x32_bf16(false, a.v, false, b.v,
                                                   (short)0, c, false, false);
}

// ---------------------------------------------------------------------------
// k0: convert concatenated weights [A_w(16 rows); M_w(256 rows)] -> bf16 [272][4096]
// ---------------------------------------------------------------------------
__global__ void k0_wcvt(const float* __restrict__ A_w, const float* __restrict__ M_w,
                        uint16_t* __restrict__ Wb) {
    int g     = blockIdx.x * blockDim.x + threadIdx.x;   // 544*256 = 272*512
    int row   = g >> 9;
    int chunk = g & 511;
    const float* src = (row < 16 ? A_w + (size_t)row * IND
                                 : M_w + (size_t)(row - 16) * IND) + chunk * 8;
    float4 f0 = *(const float4*)src;
    float4 f1 = *(const float4*)(src + 4);
    uint4 o;
    o.x = pkbf(f0.x, f0.y);  o.y = pkbf(f0.z, f0.w);
    o.z = pkbf(f1.x, f1.y);  o.w = pkbf(f1.z, f1.w);
    *(uint4*)(Wb + (size_t)row * IND + chunk * 8) = o;
}

// ---------------------------------------------------------------------------
// k1m: M-projection. grid(512,4), block(32,4). Wave wv owns M-tiles wv*4..wv*4+3
//      (uniform; wv made scalar via readfirstlane -> no EXEC games near WMMA).
//      Double-buffered 32-K sub-blocks; full unroll folds kb into load imms.
// ---------------------------------------------------------------------------
struct SubM { float4 a0, a1, a2, a3; uint4 b0[4], b1[4]; };

__device__ __forceinline__ void loadSubM(SubM& s, const float* __restrict__ xb, int xoff,
                                         const uint16_t* const* wb, int woff, int kb) {
    s.a0 = *(const float4*)(xb + xoff + kb);
    s.a1 = *(const float4*)(xb + xoff + kb + 4);
    s.a2 = *(const float4*)(xb + xoff + kb + 16);
    s.a3 = *(const float4*)(xb + xoff + kb + 20);
    #pragma unroll
    for (int t = 0; t < 4; ++t) {
        s.b0[t] = *(const uint4*)(wb[t] + woff + kb);
        s.b1[t] = *(const uint4*)(wb[t] + woff + kb + 16);
    }
}

__device__ __forceinline__ void computeSubM(const SubM& s, v8f acc[4]) {
    FragU af;
    af.u[0] = pkbf(s.a0.x, s.a0.y);  af.u[1] = pkbf(s.a0.z, s.a0.w);
    af.u[2] = pkbf(s.a1.x, s.a1.y);  af.u[3] = pkbf(s.a1.z, s.a1.w);
    af.u[4] = pkbf(s.a2.x, s.a2.y);  af.u[5] = pkbf(s.a2.z, s.a2.w);
    af.u[6] = pkbf(s.a3.x, s.a3.y);  af.u[7] = pkbf(s.a3.z, s.a3.w);
    #pragma unroll
    for (int t = 0; t < 4; ++t) {
        FragU bf;
        bf.u[0] = s.b0[t].x; bf.u[1] = s.b0[t].y; bf.u[2] = s.b0[t].z; bf.u[3] = s.b0[t].w;
        bf.u[4] = s.b1[t].x; bf.u[5] = s.b1[t].y; bf.u[6] = s.b1[t].z; bf.u[7] = s.b1[t].w;
        acc[t] = wmma_bf16(af, bf, acc[t]);
    }
}

__global__ void k1m(const float* __restrict__ x, const uint16_t* __restrict__ Wb,
                    float* __restrict__ zall) {
    const int lane   = threadIdx.x;
    const int wv     = __builtin_amdgcn_readfirstlane(threadIdx.y);  // scalar 0..3
    const int tt     = blockIdx.x;
    const int ks     = blockIdx.y;
    const int m      = lane & 15;
    const int kb_off = (lane >> 4) * 8;

    const float* xbase = x + (size_t)tt * 16 * IND + (size_t)ks * KSEG;
    const int    xoff  = m * IND + kb_off;

    const uint16_t* wb[4];
    #pragma unroll
    for (int t = 0; t < 4; ++t)   // M-tile ct = wv*4+t -> Wb rows 16 + ct*16 ..
        wb[t] = Wb + (size_t)(16 + (wv * 4 + t) * 16) * IND + (size_t)ks * KSEG;
    const int woff = m * IND + kb_off;

    v8f acc[4];
    #pragma unroll
    for (int t = 0; t < 4; ++t)
        #pragma unroll
        for (int e = 0; e < 8; ++e) acc[t][e] = 0.0f;

    SubM u, v;
    loadSubM(u, xbase, xoff, wb, woff, 0);
    #pragma unroll
    for (int kb = 0; kb < KSEG; kb += 64) {
        loadSubM(v, xbase, xoff, wb, woff, kb + 32);       // prefetch (always in seg)
        computeSubM(u, acc);
        const int nkb = (kb + 64 < KSEG) ? kb + 64 : 0;     // clamp: last pf discarded
        loadSubM(u, xbase, xoff, wb, woff, nkb);
        computeSubM(v, acc);
    }

    const int rowadd = (lane >> 4) * 8;
    #pragma unroll
    for (int t = 0; t < 4; ++t) {
        const int col = 16 + (wv * 4 + t) * 16 + m;
        #pragma unroll
        for (int r = 0; r < 8; ++r) {
            int tok = tt * 16 + r + rowadd;
            zall[((size_t)ks * TOK + tok) * NCOL + col] = acc[t][r];
        }
    }
}

// ---------------------------------------------------------------------------
// k1z: z-projection (A_w tile, cols 0..15). grid(512,4), block 32 (1 wave).
// ---------------------------------------------------------------------------
struct SubZ { float4 a0, a1, a2, a3; uint4 b0, b1; };

__device__ __forceinline__ void loadSubZ(SubZ& s, const float* __restrict__ xb, int xoff,
                                         const uint16_t* __restrict__ wb, int woff, int kb) {
    s.a0 = *(const float4*)(xb + xoff + kb);
    s.a1 = *(const float4*)(xb + xoff + kb + 4);
    s.a2 = *(const float4*)(xb + xoff + kb + 16);
    s.a3 = *(const float4*)(xb + xoff + kb + 20);
    s.b0 = *(const uint4*)(wb + woff + kb);
    s.b1 = *(const uint4*)(wb + woff + kb + 16);
}

__device__ __forceinline__ void computeSubZ(const SubZ& s, v8f& acc) {
    FragU af, bf;
    af.u[0] = pkbf(s.a0.x, s.a0.y);  af.u[1] = pkbf(s.a0.z, s.a0.w);
    af.u[2] = pkbf(s.a1.x, s.a1.y);  af.u[3] = pkbf(s.a1.z, s.a1.w);
    af.u[4] = pkbf(s.a2.x, s.a2.y);  af.u[5] = pkbf(s.a2.z, s.a2.w);
    af.u[6] = pkbf(s.a3.x, s.a3.y);  af.u[7] = pkbf(s.a3.z, s.a3.w);
    bf.u[0] = s.b0.x; bf.u[1] = s.b0.y; bf.u[2] = s.b0.z; bf.u[3] = s.b0.w;
    bf.u[4] = s.b1.x; bf.u[5] = s.b1.y; bf.u[6] = s.b1.z; bf.u[7] = s.b1.w;
    acc = wmma_bf16(af, bf, acc);
}

__global__ void k1z(const float* __restrict__ x, const uint16_t* __restrict__ Wb,
                    float* __restrict__ zall) {
    const int lane   = threadIdx.x;
    const int tt     = blockIdx.x;
    const int ks     = blockIdx.y;
    const int m      = lane & 15;
    const int kb_off = (lane >> 4) * 8;

    const float*    xbase = x  + (size_t)tt * 16 * IND + (size_t)ks * KSEG;
    const uint16_t* wbase = Wb + (size_t)ks * KSEG;      // rows 0..15 = A_w
    const int xoff = m * IND + kb_off;
    const int woff = m * IND + kb_off;

    v8f acc;
    #pragma unroll
    for (int e = 0; e < 8; ++e) acc[e] = 0.0f;

    SubZ u, v;
    loadSubZ(u, xbase, xoff, wbase, woff, 0);
    #pragma unroll
    for (int kb = 0; kb < KSEG; kb += 64) {
        loadSubZ(v, xbase, xoff, wbase, woff, kb + 32);
        computeSubZ(u, acc);
        const int nkb = (kb + 64 < KSEG) ? kb + 64 : 0;
        loadSubZ(u, xbase, xoff, wbase, woff, nkb);
        computeSubZ(v, acc);
    }

    const int rowadd = (lane >> 4) * 8;
    #pragma unroll
    for (int r = 0; r < 8; ++r) {
        int tok = tt * 16 + r + rowadd;
        zall[((size_t)ks * TOK + tok) * NCOL + m] = acc[r];
    }
}

// ---------------------------------------------------------------------------
// k2: reduce K-partials, per-token 16x16 mixing matvec, fold 1024 scale, bf16 out.
// ---------------------------------------------------------------------------
__global__ void k2_mix(const float* __restrict__ zall, const float* __restrict__ M_b,
                       uint16_t* __restrict__ zmix) {
    int g = blockIdx.x * blockDim.x + threadIdx.x;   // 131072 = T*16
    int t = g >> 4;
    int i = g & 15;
    float z[16];
    #pragma unroll
    for (int j = 0; j < 16; ++j) {
        float v = 0.0f;
        #pragma unroll
        for (int ks = 0; ks < KSPLIT; ++ks)
            v += zall[((size_t)ks * TOK + t) * NCOL + j];
        z[j] = v;
    }
    float acc = 0.0f;
    #pragma unroll
    for (int j = 0; j < 16; ++j) {
        float p = M_b[i * 16 + j];
        #pragma unroll
        for (int ks = 0; ks < KSPLIT; ++ks)
            p += zall[((size_t)ks * TOK + t) * NCOL + 16 + i * 16 + j];
        acc += p * z[j];
    }
    acc *= 1024.0f;   // NUM_HEADS (on M) * NUM_HEADS*SCALING (on out), exact pow2
    uint32_t u = __float_as_uint(acc) + 0x8000u;
    zmix[(size_t)t * 16 + i] = (uint16_t)(u >> 16);
}

// ---------------------------------------------------------------------------
// k3: out = z_mix @ B_w^T, one WMMA per 16x16 tile (K=16 zero-padded to 32).
// ---------------------------------------------------------------------------
__global__ void k3_out(const uint16_t* __restrict__ zmix, const float* __restrict__ B_w,
                       float* __restrict__ out) {
    const int lane   = threadIdx.x;
    const int wv     = __builtin_amdgcn_readfirstlane(threadIdx.y);  // scalar 0..7
    const int tt     = blockIdx.x;
    const int nt     = blockIdx.y * 8 + wv;
    const int m      = lane & 15;
    const int kb_off = (lane >> 4) * 8;

    FragU af;
    uint4 az = *(const uint4*)(zmix + (size_t)(tt * 16 + m) * 16 + kb_off);
    af.u[0] = az.x; af.u[1] = az.y; af.u[2] = az.z; af.u[3] = az.w;
    af.u[4] = 0;    af.u[5] = 0;    af.u[6] = 0;    af.u[7] = 0;

    const int o = nt * 16 + m;
    const float* brow = B_w + (size_t)o * 16 + kb_off;
    float4 b0 = *(const float4*)brow;
    float4 b1 = *(const float4*)(brow + 4);
    FragU bf;
    bf.u[0] = pkbf(b0.x, b0.y);  bf.u[1] = pkbf(b0.z, b0.w);
    bf.u[2] = pkbf(b1.x, b1.y);  bf.u[3] = pkbf(b1.z, b1.w);
    bf.u[4] = 0; bf.u[5] = 0; bf.u[6] = 0; bf.u[7] = 0;

    v8f acc;
    #pragma unroll
    for (int e = 0; e < 8; ++e) acc[e] = 0.0f;
    acc = wmma_bf16(af, bf, acc);

    const int rowadd = (lane >> 4) * 8;
    #pragma unroll
    for (int r = 0; r < 8; ++r)
        out[(size_t)(tt * 16 + r + rowadd) * OUTD + o] = acc[r];
}

// ---------------------------------------------------------------------------
// launcher: ws = Wb bf16 (2,228,224 B) | z_all fp32 x4 (35,651,584 B) | z_mix bf16
// ---------------------------------------------------------------------------
extern "C" void kernel_launch(void* const* d_in, const int* in_sizes, int n_in,
                              void* d_out, int out_size, void* d_ws, size_t ws_size,
                              hipStream_t stream) {
    const float* x   = (const float*)d_in[0];
    const float* A_w = (const float*)d_in[1];
    const float* B_w = (const float*)d_in[2];
    const float* M_w = (const float*)d_in[3];
    const float* M_b = (const float*)d_in[4];
    float* out = (float*)d_out;

    char* ws = (char*)d_ws;
    uint16_t* Wb   = (uint16_t*)(ws);
    float*    zall = (float*)(ws + 2228224);
    uint16_t* zmix = (uint16_t*)(ws + 2228224 + 35651584);

    k0_wcvt<<<544, 256, 0, stream>>>(A_w, M_w, Wb);
    k1m<<<dim3(512, 4), dim3(32, 4), 0, stream>>>(x, Wb, zall);
    k1z<<<dim3(512, 4), dim3(32, 1), 0, stream>>>(x, Wb, zall);
    k2_mix<<<512, 256, 0, stream>>>(zall, M_b, zmix);
    k3_out<<<dim3(512, 32), dim3(32, 8), 0, stream>>>(zmix, B_w, out);
}
